// KNNDistanceLoss_4844723110165
// MI455X (gfx1250) — matrix operand order repaired and checked by reading, not compiled, not run on hardware
//
#include <hip/hip_runtime.h>
#include <math.h>

// Problem constants (from reference)
#define NPTS   12288
#define DIMS   64
#define CDIM   3
#define KNN    85
#define GAMMA  0.5f
#define SAMPLE 2000
#define STILES (SAMPLE / 16)        // 125 tiles per dim
#define NTILES (STILES * STILES)    // 15625 16x16 tiles

typedef __attribute__((ext_vector_type(16))) __bf16       v16bf;
typedef __attribute__((ext_vector_type(8)))  float        v8f;
typedef __attribute__((ext_vector_type(4)))  unsigned int uint4v;

union BfVec {
    uint4v u[2];   // two 16-byte halves
    v16bf  v;      // 16 bf16 lanes for WMMA
};

// ---------------------------------------------------------------------------
// Workspace layout (byte offsets into d_ws):
//   0      : double acc[8]   (0:Σed 1:Σcd 2:Σed² 3:Σcd² 4:Σed·cd 5:Σlocal)
//   256    : float  nrm[2000]          sampled-embedding squared norms (f32)
//   8448   : float  coord_s[2000*3]    sampled coords
//   32768  : ushort embs[2000*64]      sampled embeddings in bf16 (row-major)
//   294912 : int    knn[12288*85]      neighbor indices
// ---------------------------------------------------------------------------

__global__ void k_init(double* __restrict__ acc) {
    if (threadIdx.x < 8) acc[threadIdx.x] = 0.0;
}

// Gather the 2000-point sample via a deterministic affine permutation
// (stand-in for jax.random.permutation(key(42)); gcd(4097,12288)==1 so it is
// a true permutation of [0,N)). Converts embeddings to bf16 (RNE) and
// computes f32 squared norms from the original f32 data.
__global__ void k_sample(const float* __restrict__ emb,
                         const float* __restrict__ coords,
                         unsigned short* __restrict__ embs,
                         float* __restrict__ nrm,
                         float* __restrict__ cs) {
    __shared__ float red[64];
    const int s = blockIdx.x;
    const int t = threadIdx.x;
    const int src = (int)(((long long)s * 4097LL + 1234LL) % NPTS);

    float v = emb[(size_t)src * DIMS + t];
    unsigned int u = __float_as_uint(v);
    unsigned int r = (u + 0x7FFFu + ((u >> 16) & 1u)) >> 16;   // bf16 RNE
    embs[(size_t)s * DIMS + t] = (unsigned short)r;

    red[t] = v * v;
    __syncthreads();
    for (int o = 32; o > 0; o >>= 1) {
        if (t < o) red[t] += red[t + o];
        __syncthreads();
    }
    if (t == 0) nrm[s] = red[0];
    if (t < CDIM) cs[s * CDIM + t] = coords[(size_t)src * CDIM + t];
}

// One wave per 16x16 tile of the 2000x2000 pair space. Gram tile via two
// v_wmma_f32_16x16x32_bf16 (K=64), distances reconstructed from norms,
// coordinate distances computed directly, 5 Pearson moments accumulated and
// wave-reduced, then f64 atomics into the workspace accumulators.
__global__ void k_pearson(const unsigned short* __restrict__ embs,
                          const float* __restrict__ nrm,
                          const float* __restrict__ cs,
                          double* __restrict__ acc) {
    const int wid  = threadIdx.x >> 5;
    const int lane = threadIdx.x & 31;
    const int tile = blockIdx.x * 8 + wid;
    if (tile >= NTILES) return;                 // wave-uniform: EXEC stays full

    const int tm = tile / STILES;
    const int tn = tile % STILES;
    const int hi = lane >> 4;                   // half-wave select (ISA layout)
    const int m  = lane & 15;

    const unsigned short* Arow = embs + (size_t)(tm * 16 + m) * DIMS;
    const unsigned short* Brow = embs + (size_t)(tn * 16 + m) * DIMS;

    v8f c = {};
#pragma unroll
    for (int kc = 0; kc < 2; ++kc) {
        // A 16x32 bf16: lane holds row M=lane&15; K = 8*hi+e (e<8) and
        // 16+8*hi+(e-8) (e>=8) -> two contiguous 16B loads.
        BfVec a, b;
        a.u[0] = *(const uint4v*)(Arow + 32 * kc + 8 * hi);
        a.u[1] = *(const uint4v*)(Arow + 32 * kc + 16 + 8 * hi);
        // B 32x16 bf16: lane holds col N=lane&15; K = 8*hi+e (e<8) and
        // 16+8*hi+(e-8) (e>=8) -> two contiguous 16B loads.
        b.u[0] = *(const uint4v*)(Brow + 32 * kc + 8 * hi);
        b.u[1] = *(const uint4v*)(Brow + 32 * kc + 16 + 8 * hi);
        c = __builtin_amdgcn_wmma_f32_16x16x32_bf16(
                false, a.v, false, b.v, (short)0, c, false, false);
    }

    // C/D layout: element v of lane -> M = v + 8*hi, N = lane&15
    float se = 0.f, scd = 0.f, see = 0.f, scc = 0.f, sec = 0.f;
#pragma unroll
    for (int v = 0; v < 8; ++v) {
        const int r   = tm * 16 + v + 8 * hi;
        const int col = tn * 16 + m;
        float sq = nrm[r] + nrm[col] - 2.0f * c[v];
        float ed = sq > 0.f ? sqrtf(sq) : 0.f;
        float dx = cs[r * 3 + 0] - cs[col * 3 + 0];
        float dy = cs[r * 3 + 1] - cs[col * 3 + 1];
        float dz = cs[r * 3 + 2] - cs[col * 3 + 2];
        float cq = dx * dx + dy * dy + dz * dz;
        float cd = cq > 0.f ? sqrtf(cq) : 0.f;
        se += ed; scd += cd; see += ed * ed; scc += cd * cd; sec += ed * cd;
    }
    // wave32 butterfly reduction
    for (int o = 16; o > 0; o >>= 1) {
        se  += __shfl_xor(se,  o);
        scd += __shfl_xor(scd, o);
        see += __shfl_xor(see, o);
        scc += __shfl_xor(scc, o);
        sec += __shfl_xor(sec, o);
    }
    if (lane == 0) {
        atomicAdd(&acc[0], (double)se);
        atomicAdd(&acc[1], (double)scd);
        atomicAdd(&acc[2], (double)see);
        atomicAdd(&acc[3], (double)scc);
        atomicAdd(&acc[4], (double)sec);
    }
}

// One workgroup per query point: squared distances to all N points cached in
// LDS (49KB), then 85 argmin rounds with LDS tree reduction. Selection on
// squared distance (same ordering as sqrt); self excluded by poisoning.
__global__ void k_knn(const float* __restrict__ coords, int* __restrict__ knn) {
    __shared__ float dist[NPTS];
    __shared__ float sval[256];
    __shared__ int   sidx[256];
    const int i = blockIdx.x;
    const int t = threadIdx.x;
    const float xi = coords[i * 3 + 0];
    const float yi = coords[i * 3 + 1];
    const float zi = coords[i * 3 + 2];

    for (int j = t; j < NPTS; j += 256) {
        float dx = xi - coords[j * 3 + 0];
        float dy = yi - coords[j * 3 + 1];
        float dz = zi - coords[j * 3 + 2];
        dist[j] = dx * dx + dy * dy + dz * dz;
    }
    __syncthreads();
    if (t == 0) dist[i] = 3.0e38f;              // drop self (the k+1 slot)
    __syncthreads();

    for (int k = 0; k < KNN; ++k) {
        float best = 3.0e38f;
        int   bj   = NPTS;
        for (int j = t; j < NPTS; j += 256) {
            float v = dist[j];
            if (v < best) { best = v; bj = j; }
        }
        sval[t] = best; sidx[t] = bj;
        __syncthreads();
        for (int o = 128; o > 0; o >>= 1) {
            if (t < o) {
                float ov = sval[t + o]; int oi = sidx[t + o];
                if (ov < sval[t] || (ov == sval[t] && oi < sidx[t])) {
                    sval[t] = ov; sidx[t] = oi;
                }
            }
            __syncthreads();
        }
        if (t == 0) {
            knn[(size_t)i * KNN + k] = sidx[0];
            dist[sidx[0]] = 3.0e38f;
        }
        __syncthreads();
    }
}

// One wave per point: emb_i staged in LDS, lanes stride over the 85 neighbors,
// weighted squared-error accumulated and wave-reduced into acc[5].
__global__ void k_local(const float* __restrict__ emb,
                        const float* __restrict__ coords,
                        const int* __restrict__ knn,
                        double* __restrict__ acc) {
    __shared__ float semb[8][DIMS];
    const int wid  = threadIdx.x >> 5;
    const int lane = threadIdx.x & 31;
    const int i    = blockIdx.x * 8 + wid;      // grid sized so i < NPTS always

    semb[wid][lane]      = emb[(size_t)i * DIMS + lane];
    semb[wid][lane + 32] = emb[(size_t)i * DIMS + lane + 32];
    __syncthreads();

    const float xi = coords[i * 3 + 0];
    const float yi = coords[i * 3 + 1];
    const float zi = coords[i * 3 + 2];

    float s = 0.f;
    for (int n = lane; n < KNN; n += 32) {
        const int j = knn[(size_t)i * KNN + n];
        const float* ej = emb + (size_t)j * DIMS;
        float pq = 0.f;
#pragma unroll
        for (int d = 0; d < DIMS; ++d) {
            float df = semb[wid][d] - ej[d];
            pq += df * df;
        }
        float dx = xi - coords[j * 3 + 0];
        float dy = yi - coords[j * 3 + 1];
        float dz = zi - coords[j * 3 + 2];
        float tq = dx * dx + dy * dy + dz * dz;
        float pd = pq > 0.f ? sqrtf(pq) : 0.f;
        float td = tq > 0.f ? sqrtf(tq) : 0.f;
        float diff = pd - td;
        s += diff * diff * expf(-GAMMA * td);
    }
    for (int o = 16; o > 0; o >>= 1) s += __shfl_xor(s, o);
    if (lane == 0) atomicAdd(&acc[5], (double)s);
}

__global__ void k_final(const double* __restrict__ acc, float* __restrict__ out) {
    const double M2  = (double)SAMPLE * (double)SAMPLE;
    const double me  = acc[0] / M2;
    const double mc  = acc[1] / M2;
    const double ve  = acc[2] / M2 - me * me;     // E[e^2] - E[e]^2
    const double vc  = acc[3] / M2 - mc * mc;
    const double cov = acc[4] / M2 - me * mc;     // E[ec*cc]
    const double es  = sqrt(ve + 1e-8);
    const double csd = sqrt(vc + 1e-8);
    const double pearson = cov / (es * csd + 1e-8);
    const double ploss   = 1.0 - pearson;
    const double lloss   = acc[5] / ((double)NPTS * (double)KNN);
    out[0] = (float)(ploss + 0.5 * lloss);
}

extern "C" void kernel_launch(void* const* d_in, const int* in_sizes, int n_in,
                              void* d_out, int out_size, void* d_ws, size_t ws_size,
                              hipStream_t stream) {
    const float* emb    = (const float*)d_in[0];   // (12288, 64) f32
    const float* coords = (const float*)d_in[1];   // (12288, 3)  f32
    float* out = (float*)d_out;

    char* ws = (char*)d_ws;
    double*         acc  = (double*)(ws + 0);
    float*          nrm  = (float*)(ws + 256);
    float*          cs   = (float*)(ws + 8448);
    unsigned short* embs = (unsigned short*)(ws + 32768);
    int*            knn  = (int*)(ws + 294912);

    k_init   <<<1, 32, 0, stream>>>(acc);
    k_sample <<<SAMPLE, 64, 0, stream>>>(emb, coords, embs, nrm, cs);
    k_pearson<<<(NTILES + 7) / 8, 256, 0, stream>>>(embs, nrm, cs, acc);
    k_knn    <<<NPTS, 256, 0, stream>>>(coords, knn);
    k_local  <<<NPTS / 8, 256, 0, stream>>>(emb, coords, knn, acc);
    k_final  <<<1, 1, 0, stream>>>(acc, out);
}
